// MaskTerm_32074815766904
// MI455X (gfx1250) — matrix operand order repaired and checked by reading, not compiled, not run on hardware
//
#include <hip/hip_runtime.h>
#include <hip/hip_bf16.h>

typedef __attribute__((ext_vector_type(16))) _Float16 v16h;
typedef __attribute__((ext_vector_type(8)))  float    v8f;
typedef __attribute__((ext_vector_type(4)))  float    v4f;

#define MSZ   28          // mask is 28x28
#define CH    256         // canvas H
#define CW    384         // canvas W
#define BOX_SCALE 0.25f

// ---------------------------------------------------------------------------
// Kernel 1: streaming zero-fill of the canvas (the bandwidth-bound part).
// 78.6 MB of zeros -> 128-bit non-temporal stores, grid-stride.
// ---------------------------------------------------------------------------
__global__ __launch_bounds__(256) void zero_fill_kernel(float* __restrict__ out,
                                                        int n4, int ntotal) {
    long i      = (long)blockIdx.x * blockDim.x + threadIdx.x;
    long stride = (long)gridDim.x * blockDim.x;
    v4f z = {0.0f, 0.0f, 0.0f, 0.0f};
    v4f* __restrict__ o4 = (v4f*)out;
    for (long k = i; k < n4; k += stride)
        __builtin_nontemporal_store(z, &o4[k]);
    // tail (out_size is 4-divisible here, but be safe)
    int tail = ntotal - (n4 << 2);
    if (blockIdx.x == 0 && (int)threadIdx.x < tail)
        out[(n4 << 2) + threadIdx.x] = 0.0f;
}

// ---------------------------------------------------------------------------
// Kernel 2: per-box bilinear resize as two GEMMs via v_wmma_f32_16x16x32_f16.
//   O(h x w) = Ry(h x 28) * mask(28 x 28) * Rx(28 x w)
// One block per box, 8 waves; wave w owns output row-tile [16w, 16w+16).
// ---------------------------------------------------------------------------
__global__ __launch_bounds__(256) void paste_wmma_kernel(
    const float* __restrict__ masks,   // [N,1,28,28] f32
    const float* __restrict__ boxes,   // [N,5] f32: [0, x0, y0, x1, y1]
    float* __restrict__ out)           // [1,N,256,384] f32 (pre-zeroed)
{
    __shared__ _Float16 lds_mask[32 * MSZ];   // rows K=0..31 (28..31 zero), cols x=0..27
    __shared__ _Float16 lds_t[8][16 * 32];    // per-wave T strip: 16 rows x 32 K (f16)

    const int n    = blockIdx.x;
    const int tid  = threadIdx.x;
    const int lane = tid & 31;
    const int wid  = tid >> 5;

    // ---- box geometry (uniform per block) ----
    const int x0c = (int)(boxes[n * 5 + 1] * BOX_SCALE);
    const int y0c = (int)(boxes[n * 5 + 2] * BOX_SCALE);
    const int x1c = (int)(boxes[n * 5 + 3] * BOX_SCALE);
    const int y1c = (int)(boxes[n * 5 + 4] * BOX_SCALE);
    int wbox = x1c - x0c + 1; if (wbox < 1) wbox = 1;
    int hbox = y1c - y0c + 1; if (hbox < 1) hbox = 1;
    const float sx = (float)MSZ / (float)wbox;
    const float sy = (float)MSZ / (float)hbox;

    // ---- stage mask into LDS as f16, K rows padded to 32 with zeros ----
    for (int i = tid; i < 32 * MSZ; i += 256) {
        int k = i / MSZ, x = i - k * MSZ;
        float v = (k < MSZ) ? masks[(size_t)n * (MSZ * MSZ) + k * MSZ + x] : 0.0f;
        lds_mask[i] = (_Float16)v;
    }
    __syncthreads();

    const int rt  = wid;                       // this wave's 16-row tile of the box
    const int nrt = (hbox + 15) >> 4;          // <= 7 since hbox <= 101
    if (rt >= nrt) return;                     // wave-uniform exit (no barriers below)

    // fragment-layout helpers (CDNA5 ISA, 16-bit A 16x32 / B 32x16 / f32 C 16x16)
    const int ncol = lane & 15;                // B/D column, A row (mod 16)
    const int kofsB = (lane < 16) ? 0 : 16;    // unused; B fragments take K from kloA/khiA, same order as A
    const int kloA  = (lane < 16) ? 0 : 8;     // A: VGPR0-3 K-base
    const int khiA  = (lane < 16) ? 16 : 24;   // A: VGPR4-7 K-base
    const int mofsD = (lane < 16) ? 0 : 8;     // C/D: row = v + mofsD

    // ---- stage 1 A fragment: row-interp weights Ry[r][K] built in registers ----
    const int   r     = rt * 16 + ncol;        // box-relative row this lane represents
    const float src_y = fmaxf(((float)r + 0.5f) * sy - 0.5f, 0.0f);
    int   ylo = (int)floorf(src_y); if (ylo > MSZ - 1) ylo = MSZ - 1;
    int   yhi = ylo + 1;            if (yhi > MSZ - 1) yhi = MSZ - 1;
    const float wy = src_y - (float)ylo;

    v16h afrag;
    #pragma unroll
    for (int t = 0; t < 16; ++t) {
        int v = t >> 1, s = t & 1;
        int K = (v < 4) ? (kloA + 2 * v + s) : (khiA + 2 * (v - 4) + s);
        float a = 0.0f;
        if (K == ylo) a += 1.0f - wy;
        if (K == yhi) a += wy;
        afrag[t] = (_Float16)a;
    }

    // ---- stage 1 B fragments: mask columns [0..15] and [16..31] (cols>=28 are zero) ----
    v16h b0, b1;
    #pragma unroll
    for (int t = 0; t < 16; ++t) {
        int K  = (t < 8) ? (kloA + t) : (khiA + t - 8); // 0..31; rows >= 28 are zero-padded in LDS
        int c1 = 16 + ncol;
        b0[t] = lds_mask[K * MSZ + ncol];
        b1[t] = (c1 < MSZ) ? lds_mask[K * MSZ + c1] : (_Float16)0.0f;
    }

    v8f t0 = {}, t1 = {};
    t0 = __builtin_amdgcn_wmma_f32_16x16x32_f16(false, afrag, false, b0, (short)0, t0, false, false);
    t1 = __builtin_amdgcn_wmma_f32_16x16x32_f16(false, afrag, false, b1, (short)0, t1, false, false);

    // ---- D-layout -> A-layout via wave-private LDS strip (cols 28..31 are exact zeros) ----
    _Float16* tstrip = &lds_t[wid][0];
    #pragma unroll
    for (int v = 0; v < 8; ++v) {
        tstrip[(v + mofsD) * 32 + ncol]      = (_Float16)t0[v];
        tstrip[(v + mofsD) * 32 + 16 + ncol] = (_Float16)t1[v];
    }
    asm volatile("s_wait_dscnt 0" ::: "memory");   // DS in-order per wave; force the wait

    v16h a2;
    #pragma unroll
    for (int t = 0; t < 16; ++t) {
        int v = t >> 1, s = t & 1;
        int K = (v < 4) ? (kloA + 2 * v + s) : (khiA + 2 * (v - 4) + s);
        a2[t] = tstrip[ncol * 32 + K];
    }

    // ---- stage 2: O tile = T * Rx, one WMMA per 16-col tile; predicated stores ----
    const int nct = (wbox + 15) >> 4;          // <= 7
    const size_t outbase = (size_t)n * (CH * CW);
    for (int ct = 0; ct < nct; ++ct) {
        const int   c     = ct * 16 + ncol;    // box-relative column this lane represents
        const float src_x = fmaxf(((float)c + 0.5f) * sx - 0.5f, 0.0f);
        int xlo = (int)floorf(src_x); if (xlo > MSZ - 1) xlo = MSZ - 1;
        int xhi = xlo + 1;            if (xhi > MSZ - 1) xhi = MSZ - 1;
        const float wx = src_x - (float)xlo;

        v16h b2;
        #pragma unroll
        for (int t = 0; t < 16; ++t) {
            int K = (t < 8) ? (kloA + t) : (khiA + t - 8);
            float b = 0.0f;
            if (K == xlo) b += 1.0f - wx;
            if (K == xhi) b += wx;
            b2[t] = (_Float16)b;
        }

        v8f acc = {};
        acc = __builtin_amdgcn_wmma_f32_16x16x32_f16(false, a2, false, b2, (short)0, acc, false, false);

        #pragma unroll
        for (int v = 0; v < 8; ++v) {
            int rr = rt * 16 + v + mofsD;      // box-relative row
            int cc = ct * 16 + ncol;           // box-relative col
            if (rr < hbox && cc < wbox) {
                int Y = y0c + rr, X = x0c + cc;
                if (Y >= 0 && Y < CH && X >= 0 && X < CW)
                    out[outbase + (size_t)Y * CW + X] = acc[v];
            }
        }
    }
}

// ---------------------------------------------------------------------------
extern "C" void kernel_launch(void* const* d_in, const int* in_sizes, int n_in,
                              void* d_out, int out_size, void* d_ws, size_t ws_size,
                              hipStream_t stream) {
    const float* masks = (const float*)d_in[0];   // (200,1,28,28) f32
    const float* boxes = (const float*)d_in[1];   // (200,5)       f32
    // d_in[2] (cls_indices) and d_in[3] (seg_score) are unused by the reference output
    float* out = (float*)d_out;                    // (1,200,256,384) f32

    const int n4 = out_size >> 2;
    zero_fill_kernel<<<2048, 256, 0, stream>>>(out, n4, out_size);

    const int nboxes = in_sizes[1] / 5;            // 200
    paste_wmma_kernel<<<nboxes, 256, 0, stream>>>(masks, boxes, out);
}
